// PolyGAN_CP_Layer_73297911873805
// MI455X (gfx1250) — matrix operand order repaired and checked by reading, not compiled, not run on hardware
//
#include <hip/hip_runtime.h>
#include <hip/hip_bf16.h>

typedef __attribute__((ext_vector_type(16))) _Float16 v16h;
typedef __attribute__((ext_vector_type(8)))  float    v8f;
typedef __attribute__((ext_vector_type(4)))  float    f4;

#define NB1 512          // blocks for the reduction pass
#define T1  256          // threads per block (8 wave32)

// ---------------------------------------------------------------------------
// Kernel 1: partial dot products  d[k][r] += z[s] * W[k+1][s][r],  k=0..2
// Each block writes 12 deterministic partials to ws.  W rows are (s,4) f32 so
// each row is one b128 load; z is loaded as float4 (4 s per iteration).
// ---------------------------------------------------------------------------
__global__ __launch_bounds__(T1) void k_partial(const float* __restrict__ z,
                                                const float* __restrict__ W,
                                                float* __restrict__ part,
                                                int S) {
    const int tid    = threadIdx.x;
    const int gid    = blockIdx.x * T1 + tid;
    const int nthr   = NB1 * T1;
    const int S4     = S >> 2;

    const f4* z4 = reinterpret_cast<const f4*>(z);
    // W is (20, S, 4); factors 1..3 start at offset S*4.
    const f4* W1 = reinterpret_cast<const f4*>(W + (size_t)S * 4 * 1);
    const f4* W2 = reinterpret_cast<const f4*>(W + (size_t)S * 4 * 2);
    const f4* W3 = reinterpret_cast<const f4*>(W + (size_t)S * 4 * 3);

    float acc[12];
#pragma unroll
    for (int v = 0; v < 12; ++v) acc[v] = 0.0f;

    for (int i = gid; i < S4; i += nthr) {
        f4 zv = z4[i];
#pragma unroll
        for (int j = 0; j < 4; ++j) {
            const int s = 4 * i + j;
            const float zs = zv[j];
            f4 w1 = W1[s], w2 = W2[s], w3 = W3[s];
#pragma unroll
            for (int r = 0; r < 4; ++r) {
                acc[0 + r] = fmaf(zs, w1[r], acc[0 + r]);
                acc[4 + r] = fmaf(zs, w2[r], acc[4 + r]);
                acc[8 + r] = fmaf(zs, w3[r], acc[8 + r]);
            }
        }
    }

    // wave32 tree reduction
#pragma unroll
    for (int v = 0; v < 12; ++v) {
#pragma unroll
        for (int off = 16; off > 0; off >>= 1)
            acc[v] += __shfl_down(acc[v], off, 32);
    }

    __shared__ float wsum[T1 / 32][12];
    const int wave = tid >> 5;
    const int lane = tid & 31;
    if (lane == 0) {
#pragma unroll
        for (int v = 0; v < 12; ++v) wsum[wave][v] = acc[v];
    }
    __syncthreads();
    if (tid < 12) {
        float s = 0.0f;
#pragma unroll
        for (int w = 0; w < T1 / 32; ++w) s += wsum[w][tid];
        part[blockIdx.x * 12 + tid] = s;
    }
}

// ---------------------------------------------------------------------------
// Kernel 2: reduce the NB1 partials deterministically, form coef[0..3].
// coef[r] = 2 + d0 + d0*d1 + d0*d1*d2
// ---------------------------------------------------------------------------
__global__ __launch_bounds__(256) void k_coef(const float* __restrict__ part,
                                              float* __restrict__ coef) {
    __shared__ float red[256];
    __shared__ float dsh[12];
    const int tid = threadIdx.x;

    for (int v = 0; v < 12; ++v) {
        float s = 0.0f;
        for (int i = tid; i < NB1; i += 256) s += part[i * 12 + v];
        red[tid] = s;
        __syncthreads();
        for (int off = 128; off > 0; off >>= 1) {
            if (tid < off) red[tid] += red[tid + off];
            __syncthreads();
        }
        if (tid == 0) dsh[v] = red[0];
        __syncthreads();
    }
    if (tid < 4) {
        const float d0 = dsh[tid], d1 = dsh[4 + tid], d2 = dsh[8 + tid];
        coef[tid] = 2.0f + d0 * (1.0f + d1 * (1.0f + d2));
    }
}

// ---------------------------------------------------------------------------
// Kernel 3: out[s] = dot(W0[s], coef) + b[s], via V_WMMA_F32_16X16X32_F16
// with hi/lo f16 splitting for f32-level accuracy.
//
// Tile: one wave covers 128 consecutive s rows, s = s_base + m*8 + g
// (m = 0..15 -> WMMA M, g = 0..7 encoded in the K dimension).
//   A[m, 4g+r]   = W0[s_base + m*8 + g][r]
//   B[4g+r, n]   = coef[r] * (g == n&7)
//   D[m, n]      = dot(W0[s_base + m*8 + (n&7)], coef)     (n and n+8 dup)
// Lo residuals of W0 (~7e-7) would be f16 denormals, so A_lo is pre-scaled
// by 2048 into the normal range and the correction is rescaled at store.
// ---------------------------------------------------------------------------
__global__ __launch_bounds__(256) void k_out(const float* __restrict__ W0,
                                             const float* __restrict__ bias,
                                             const float* __restrict__ coef4,
                                             float* __restrict__ out,
                                             int S) {
    const int tid     = blockIdx.x * blockDim.x + threadIdx.x;
    const int waveId  = tid >> 5;
    const int lane    = threadIdx.x & 31;
    const int s_base  = waveId * 128;
    if (s_base + 128 > S) return;        // wave-uniform guard, before WMMA

    const int n       = lane & 15;       // column of B / D
    const int hiHalf  = lane >> 4;       // 0: K 0..15, 1: K 16..31
    const int m       = n;               // A row index for this lane
    const int g0      = hiHalf * 2;

    // coef hi/lo split (coef ~ O(1): lo is f16-normal, no scaling needed)
    _Float16 ch[4], cl[4];
#pragma unroll
    for (int r = 0; r < 4; ++r) {
        const float c  = coef4[r];
        ch[r]          = (_Float16)c;
        cl[r]          = (_Float16)(c - (float)ch[r]);
    }

    // Build block-diagonal B_hi / B_lo
    v16h Bh = {}, Bl = {};
#pragma unroll
    for (int i = 0; i < 16; ++i) {
        const int k = hiHalf * 16 + i;
        const int g = k >> 2;
        const int r = k & 3;
        if (g == (n & 7)) { Bh[i] = ch[r]; Bl[i] = cl[r]; }
    }

    // Load 4 W0 rows per lane (each a b128), split hi/lo into A
    const f4* W04 = reinterpret_cast<const f4*>(W0);
    v16h Ah = {}, Al = {};
    const int gsel[4] = { 2 * g0, 2 * g0 + 1, 2 * g0 + 2, 2 * g0 + 3 };
#pragma unroll
    for (int q = 0; q < 4; ++q) {
        const int s = s_base + m * 8 + gsel[q];
        f4 w = W04[s];
#pragma unroll
        for (int r = 0; r < 4; ++r) {
            const float x  = w[r];
            const _Float16 hx = (_Float16)x;
            Ah[q * 4 + r] = hx;
            Al[q * 4 + r] = (_Float16)((x - (float)hx) * 2048.0f);
        }
    }

    // D = Ah*Bl + Ah*Bh  ;  Dcorr = (Al*2048)*Bh
    v8f zero = {};
    v8f c1 = __builtin_amdgcn_wmma_f32_16x16x32_f16(false, Ah, false, Bl,
                                                    (short)0, zero, false, false);
    c1     = __builtin_amdgcn_wmma_f32_16x16x32_f16(false, Ah, false, Bh,
                                                    (short)0, c1,   false, false);
    v8f c2 = __builtin_amdgcn_wmma_f32_16x16x32_f16(false, Al, false, Bh,
                                                    (short)0, zero, false, false);

    // D layout: lane<16 -> M=j,   N=lane ; lane>=16 -> M=j+8, N=lane-16.
    // Columns n>=8 duplicate n-8; only n<8 stores.  Per VGPR j the 8 active
    // lanes store 8 contiguous floats.
    if (n < 8) {
#pragma unroll
        for (int j = 0; j < 8; ++j) {
            const int mo = j + hiHalf * 8;
            const int s  = s_base + mo * 8 + n;
            out[s] = c1[j] + c2[j] * (1.0f / 2048.0f) + bias[s];
        }
    }
}

// ---------------------------------------------------------------------------
extern "C" void kernel_launch(void* const* d_in, const int* in_sizes, int n_in,
                              void* d_out, int out_size, void* d_ws, size_t ws_size,
                              hipStream_t stream) {
    const float* z = (const float*)d_in[0];
    const float* W = (const float*)d_in[1];
    const float* b = (const float*)d_in[2];
    float* out = (float*)d_out;
    float* ws  = (float*)d_ws;

    const int S = in_sizes[0];

    float* part = ws;                 // NB1 * 12 floats
    float* coef = ws + NB1 * 12;      // 4 floats

    k_partial<<<NB1, T1, 0, stream>>>(z, W, part, S);
    k_coef<<<1, 256, 0, stream>>>(part, coef);

    const int waves  = S / 128;                 // 128 outputs per wave
    const int blocks = (waves + 7) / 8;         // 8 waves per 256-thread block
    k_out<<<blocks, 256, 0, stream>>>(W /* = W[0] */, b, coef, out, S);
}
